// Encoder_14345190768997
// MI455X (gfx1250) — compile-verified, matrix operand order checked by reading, not run on hardware
//
#include <hip/hip_runtime.h>
#include <hip/hip_bf16.h>

typedef __attribute__((ext_vector_type(16))) _Float16 v16h;
typedef __attribute__((ext_vector_type(8)))  _Float16 v8h;
typedef __attribute__((ext_vector_type(8)))  float    v8f;

#define IN_CH 128
#define HID   64

// ---------------- elementwise helpers ----------------

__global__ void gcn_cvt_f16(const float* __restrict__ in, _Float16* __restrict__ out, int n) {
    int i = blockIdx.x * blockDim.x + threadIdx.x;
    if (i < n) out[i] = (_Float16)in[i];
}

__global__ void gcn_set_const(float* __restrict__ p, float v, int n) {
    int i = blockIdx.x * blockDim.x + threadIdx.x;
    if (i < n) p[i] = v;
}

__global__ void gcn_out_init_bias(float* __restrict__ out, const float* __restrict__ b, int n) {
    int i = blockIdx.x * blockDim.x + threadIdx.x;
    if (i < n) out[i] = b[i & (HID - 1)];
}

__global__ void gcn_bias_relu_cvt(const float* __restrict__ agg, const float* __restrict__ b,
                                  _Float16* __restrict__ out, int n) {
    int i = blockIdx.x * blockDim.x + threadIdx.x;
    if (i < n) out[i] = (_Float16)fmaxf(agg[i] + b[i & (HID - 1)], 0.0f);
}

// ---------------- degree / normalization ----------------

__global__ void gcn_deg_accum(const int* __restrict__ dst, float* __restrict__ deg, int E) {
    int e = blockIdx.x * blockDim.x + threadIdx.x;
    if (e < E) atomicAdd(&deg[dst[e]], 1.0f);
}

__global__ void gcn_deg_rsqrt(float* __restrict__ deg, int n) {
    int i = blockIdx.x * blockDim.x + threadIdx.x;
    if (i < n) deg[i] = rsqrtf(deg[i]);   // deg >= 1 always (self-loop), matches reference
}

// ---------------- weight packing into WMMA B-fragment order ----------------
// B tile is 32x16 (KxN), f16. Assumed mirror of the documented A layout:
// lane l: N = nt*16 + (l&15), hi = l>>4;
// element e (0..15): K = kt*32 + hi*8 + (e&7) + (e&8 ? 16 : 0)
// Packed index: ((kt*4 + nt)*32 + lane)*16 + e   -> one contiguous v16h per lane.
__global__ void gcn_pack_w(const float* __restrict__ W, _Float16* __restrict__ Bp,
                           int K, int total) {
    int idx = blockIdx.x * blockDim.x + threadIdx.x;
    if (idx >= total) return;
    int e    = idx & 15;
    int lane = (idx >> 4) & 31;
    int tile = idx >> 9;
    int nt   = tile & 3;
    int kt   = tile >> 2;
    int hi   = lane >> 4;
    int k    = kt * 32 + hi * 8 + (e & 7) + ((e & 8) ? 16 : 0);
    int n    = nt * 16 + (lane & 15);
    Bp[idx] = (_Float16)W[k * HID + n];
    (void)K;
}

// ---------------- WMMA GEMM: C[M x 64] = A[M x K] (f16) * Bp (packed f16) ----------------
// One block = 16 output rows, 4 waves = 4 column tiles of 16. K in {128, 64}.
__global__ void __launch_bounds__(128)
gcn_gemm_wmma(const _Float16* __restrict__ A, const _Float16* __restrict__ Bp,
              float* __restrict__ C, int K) {
    const int lane = threadIdx.x & 31;
    const int nT   = threadIdx.x >> 5;          // wave id -> N tile (0..3)
    const int m0   = blockIdx.x << 4;
    const int hi   = lane >> 4;                 // lane half selects K sub-block
    const int row  = m0 + (lane & 15);
    const int kTiles = K >> 5;

    const _Float16* arow = A + (size_t)row * K + hi * 8;
    const _Float16* bp   = Bp + ((size_t)nT * 32 + lane) * 16;

    v8f acc = {0.f, 0.f, 0.f, 0.f, 0.f, 0.f, 0.f, 0.f};

    for (int kt = 0; kt < kTiles; ++kt) {
        union { v16h v; v8h h[2]; } a;
        a.h[0] = *(const v8h*)(arow);           // K = kt*32 + hi*8 + {0..7}
        a.h[1] = *(const v8h*)(arow + 16);      // K = kt*32 + hi*8 + 16 + {0..7}
        v16h b = *(const v16h*)(bp);
        __builtin_prefetch(arow + 32, 0, 0);    // next K chunk -> global_prefetch_b8
        acc = __builtin_amdgcn_wmma_f32_16x16x32_f16(
            /*neg_a=*/false, a.v, /*neg_b=*/false, b,
            /*c_mod=*/(short)0, acc, /*reuse_a=*/false, /*reuse_b=*/false);
        arow += 32;
        bp   += 4 * 32 * 16;                    // next kt group in packed layout
    }

    // C/D layout: VGPR r -> row m0 + r + 8*hi, col = nT*16 + (lane&15)
    float* crow = C + (size_t)(m0 + hi * 8) * HID + (nT << 4) + (lane & 15);
#pragma unroll
    for (int r = 0; r < 8; ++r) crow[(size_t)r * HID] = acc[r];
}

// ---------------- edge aggregation: out[dst] += H[src] * dis[src]*dis[dst] ----------------
// One wave per edge (incl. self-loops as edges E..E+N-1); lane covers 2 channels.
__global__ void __launch_bounds__(256)
gcn_aggregate(const float* __restrict__ H, const int* __restrict__ src,
              const int* __restrict__ dst, const float* __restrict__ dis,
              float* __restrict__ out, int E, int N) {
    int wave = (blockIdx.x * blockDim.x + threadIdx.x) >> 5;
    int lane = threadIdx.x & 31;
    if (wave >= E + N) return;
    int s, d;
    if (wave < E) { s = src[wave]; d = dst[wave]; }
    else          { s = d = wave - E; }
    float norm = dis[s] * dis[d];
    const float2 h2 = *(const float2*)(H + (size_t)s * HID + lane * 2);
    float* o = out + (size_t)d * HID + lane * 2;
    atomicAdd(o,     h2.x * norm);
    atomicAdd(o + 1, h2.y * norm);
}

// ---------------- launcher ----------------

extern "C" void kernel_launch(void* const* d_in, const int* in_sizes, int n_in,
                              void* d_out, int out_size, void* d_ws, size_t ws_size,
                              hipStream_t stream) {
    const float* x  = (const float*)d_in[0];
    const int*   ei = (const int*)d_in[1];
    const float* W1 = (const float*)d_in[2];
    const float* b1 = (const float*)d_in[3];
    const float* W2 = (const float*)d_in[4];
    const float* b2 = (const float*)d_in[5];
    float* out = (float*)d_out;

    const int N = in_sizes[0] / IN_CH;     // 50000 (divisible by 16)
    const int E = in_sizes[1] / 2;         // 800000
    const int* src = ei;
    const int* dst = ei + E;

    // workspace carve-up (256B aligned)
    char* ws = (char*)d_ws;
    size_t off = 0;
    auto alloc = [&](size_t bytes) -> char* {
        char* p = ws + off;
        off = (off + bytes + 255) & ~(size_t)255;
        return p;
    };
    _Float16* xh   = (_Float16*)alloc((size_t)N * IN_CH * sizeof(_Float16));
    _Float16* w1p  = (_Float16*)alloc((size_t)(IN_CH/32) * 4 * 32 * 16 * sizeof(_Float16));
    _Float16* w2p  = (_Float16*)alloc((size_t)(HID/32)   * 4 * 32 * 16 * sizeof(_Float16));
    float*    proj = (float*)   alloc((size_t)N * HID * sizeof(float));
    float*    agg  = (float*)   alloc((size_t)N * HID * sizeof(float));
    _Float16* h1h  = (_Float16*)alloc((size_t)N * HID * sizeof(_Float16));
    float*    dis  = (float*)   alloc((size_t)N * sizeof(float));

    const int T = 256;
    int t;

    // x -> f16
    t = N * IN_CH;
    gcn_cvt_f16<<<(t + T - 1) / T, T, 0, stream>>>(x, xh, t);

    // pack weights into WMMA fragment order
    t = (IN_CH / 32) * 4 * 32 * 16;
    gcn_pack_w<<<(t + T - 1) / T, T, 0, stream>>>(W1, w1p, IN_CH, t);
    t = (HID / 32) * 4 * 32 * 16;
    gcn_pack_w<<<(t + T - 1) / T, T, 0, stream>>>(W2, w2p, HID, t);

    // degree -> deg^{-1/2}  (self-loop contributes the initial 1.0)
    gcn_set_const<<<(N + T - 1) / T, T, 0, stream>>>(dis, 1.0f, N);
    gcn_deg_accum<<<(E + T - 1) / T, T, 0, stream>>>(dst, dis, E);
    gcn_deg_rsqrt<<<(N + T - 1) / T, T, 0, stream>>>(dis, N);

    // layer 1: proj = x @ W1
    gcn_gemm_wmma<<<N / 16, 128, 0, stream>>>(xh, w1p, proj, IN_CH);

    // aggregate into agg (zeroed first; workspace is poisoned by harness)
    t = N * HID;
    gcn_set_const<<<(t + T - 1) / T, T, 0, stream>>>(agg, 0.0f, t);
    {
        long long threads = (long long)(E + N) * 32;
        int blocks = (int)((threads + T - 1) / T);
        gcn_aggregate<<<blocks, T, 0, stream>>>(proj, src, dst, dis, agg, E, N);
    }

    // h1 = relu(agg + b1) -> f16
    gcn_bias_relu_cvt<<<(t + T - 1) / T, T, 0, stream>>>(agg, b1, h1h, t);

    // layer 2: proj = h1 @ W2
    gcn_gemm_wmma<<<N / 16, 128, 0, stream>>>(h1h, w2p, proj, HID);

    // out = b2, then aggregate layer-2 messages into it
    gcn_out_init_bias<<<(t + T - 1) / T, T, 0, stream>>>(out, b2, t);
    {
        long long threads = (long long)(E + N) * 32;
        int blocks = (int)((threads + T - 1) / T);
        gcn_aggregate<<<blocks, T, 0, stream>>>(proj, src, dst, dis, out, E, N);
    }
}
